// MF2Net_78451872628771
// MI455X (gfx1250) — matrix operand order repaired and checked by reading, not compiled, not run on hardware
//
#include <hip/hip_runtime.h>

typedef __bf16 bf16x16 __attribute__((ext_vector_type(16)));
typedef __bf16 bf16x2 __attribute__((ext_vector_type(2)));
typedef float f32x8 __attribute__((ext_vector_type(8)));

__device__ __forceinline__ unsigned short f2bf(float f) {
  unsigned u = __float_as_uint(f);
  u += 0x7FFFu + ((u >> 16) & 1u);           // round-to-nearest-even
  return (unsigned short)(u >> 16);
}
__device__ __forceinline__ float bf2f(unsigned short h) {
  return __uint_as_float(((unsigned)h) << 16);
}
__device__ __forceinline__ unsigned pk2(float a, float b) {
#if __has_builtin(__builtin_amdgcn_cvt_pk_bf16_f32)
  bf16x2 p = __builtin_amdgcn_cvt_pk_bf16_f32(a, b);
  union { bf16x2 v; unsigned u; } cvt;
  cvt.v = p;
  return cvt.u;
#else
  return (unsigned)f2bf(a) | ((unsigned)f2bf(b) << 16);
#endif
}

union FragU { uint4 q[2]; bf16x16 v; };

__device__ __forceinline__ f32x8 wmma_bf16(bf16x16 a, bf16x16 b, f32x8 c) {
  return __builtin_amdgcn_wmma_f32_16x16x32_bf16(false, a, false, b, (short)0, c,
                                                 false, false);
}

// 4 x 16B async global->LDS copies (64B per lane), ASYNCcnt-tracked.
__device__ __forceinline__ void async_copy64(const unsigned short* g,
                                             unsigned short* l) {
  unsigned lds = (unsigned)(unsigned long long)(const void*)l;
  unsigned long long ga = (unsigned long long)(const void*)g;
  asm volatile(
      "global_load_async_to_lds_b128 %0, %1, off\n"
      "global_load_async_to_lds_b128 %0, %1, off offset:16\n"
      "global_load_async_to_lds_b128 %0, %1, off offset:32\n"
      "global_load_async_to_lds_b128 %0, %1, off offset:48\n" ::"v"(lds),
      "v"(ga)
      : "memory");
}
// 8 x 16B async global->LDS copies (128B per lane).
__device__ __forceinline__ void async_copy128(const unsigned short* g,
                                              unsigned short* l) {
  unsigned lds = (unsigned)(unsigned long long)(const void*)l;
  unsigned long long ga = (unsigned long long)(const void*)g;
  asm volatile(
      "global_load_async_to_lds_b128 %0, %1, off\n"
      "global_load_async_to_lds_b128 %0, %1, off offset:16\n"
      "global_load_async_to_lds_b128 %0, %1, off offset:32\n"
      "global_load_async_to_lds_b128 %0, %1, off offset:48\n"
      "global_load_async_to_lds_b128 %0, %1, off offset:64\n"
      "global_load_async_to_lds_b128 %0, %1, off offset:80\n"
      "global_load_async_to_lds_b128 %0, %1, off offset:96\n"
      "global_load_async_to_lds_b128 %0, %1, off offset:112\n" ::"v"(lds),
      "v"(ga)
      : "memory");
}
__device__ __forceinline__ void wait_async4() {
  asm volatile("s_wait_asynccnt 0x4" ::: "memory");
}
__device__ __forceinline__ void wait_async0() {
  asm volatile("s_wait_asynccnt 0x0" ::: "memory");
}

// ---------------------------------------------------------------------------
// K0: W1 (4096x128 fp32, k-major) -> bf16 n-major W1t[128][4096]
// ---------------------------------------------------------------------------
__global__ void __launch_bounds__(256) wconv_kernel(
    const float* __restrict__ We, const float* __restrict__ Wa,
    unsigned short* __restrict__ Oe, unsigned short* __restrict__ Oa) {
  const float* W = blockIdx.y ? Wa : We;
  unsigned short* O = blockIdx.y ? Oa : Oe;
  int i = blockIdx.x * 256 + threadIdx.x;   // i = n*4096 + k
  int n = i >> 12;
  int k = i & 4095;
  O[i] = f2bf(W[k * 128 + n]);
}

// ---------------------------------------------------------------------------
// K1: E = relu(X @ W1 + b1)   X:16384x4096 fp32, W1t bf16 n-major, E bf16
// Block tile: 64(M) x 128(N), K-tile 64, double-buffered LDS.
// A staged via regs (fp32->bf16 convert); B staged via async-to-LDS.
// Branch-free main loop (K index wraps) so next-tile loads hoist above
// the barrier/WMMA section.
// ---------------------------------------------------------------------------
__global__ void __launch_bounds__(256) gemm1_kernel(
    const float* __restrict__ Xe, const float* __restrict__ Xa,
    const unsigned short* __restrict__ Wte, const unsigned short* __restrict__ Wta,
    const float* __restrict__ be, const float* __restrict__ ba,
    unsigned short* __restrict__ Ee, unsigned short* __restrict__ Ea) {
  const float* X = blockIdx.y ? Xa : Xe;
  const unsigned short* Wt = blockIdx.y ? Wta : Wte;
  const float* bias = blockIdx.y ? ba : be;
  unsigned short* E = blockIdx.y ? Ea : Ee;

  __shared__ unsigned short As[2][64 * 64];    // [row][k] bf16, 8KB each
  __shared__ unsigned short Bs[2][128 * 64];   // [n][k]   bf16, 16KB each

  const int tid = threadIdx.x;
  const int lane = tid & 31;
  const int wave = tid >> 5;
  const int wm = wave & 3;                  // M sub-tile (16 rows)
  const int wn = wave >> 2;                 // N sub-tile (64 cols)
  const int lr = lane & 15;
  const int sel = lane >> 4;
  const int m0 = blockIdx.x * 64;

  // staging coords
  const int arow = tid >> 2;                // 0..63
  const int aq = tid & 3;                   // 16 k each
  const int brow = tid >> 1;                // 0..127
  const int bh = tid & 1;                   // 32 k each

  f32x8 acc[4];
  const f32x8 vzero = {0.f, 0.f, 0.f, 0.f, 0.f, 0.f, 0.f, 0.f};
#pragma unroll
  for (int j = 0; j < 4; ++j) acc[j] = vzero;

  const size_t xbase = (size_t)(m0 + arow) * 4096;
  const unsigned short* wbase = Wt + (size_t)brow * 4096 + bh * 32;

  // ---- prologue: tile 0 ----
  float4 fa[4];
  {
    const float4* src = (const float4*)(X + xbase + aq * 16);
    fa[0] = src[0]; fa[1] = src[1]; fa[2] = src[2]; fa[3] = src[3];
  }
  async_copy64(wbase, &Bs[0][brow * 64 + bh * 32]);

  for (int kt = 0; kt < 64; ++kt) {
    const int cur = kt & 1;
    // convert + store current A tile
    unsigned* dA = (unsigned*)&As[cur][arow * 64 + aq * 16];
#pragma unroll
    for (int i = 0; i < 4; ++i) {
      dA[i * 2 + 0] = pk2(fa[i].x, fa[i].y);
      dA[i * 2 + 1] = pk2(fa[i].z, fa[i].w);
    }
    // issue next-tile loads unconditionally (wrapped index on last iter)
    {
      const int k0n = ((kt + 1) & 63) * 64;
      const float4* src = (const float4*)(X + xbase + k0n + aq * 16);
      fa[0] = src[0]; fa[1] = src[1]; fa[2] = src[2]; fa[3] = src[3];
      async_copy64(wbase + k0n, &Bs[1 - cur][brow * 64 + bh * 32]);
      __builtin_prefetch(
          (const void*)(X + xbase + (((kt + 2) & 63) * 64) + aq * 16), 0, 1);
    }
    wait_async4();                          // current B tile landed
    __syncthreads();

#pragma unroll
    for (int ks = 0; ks < 2; ++ks) {
      FragU a;
      const int row = 16 * wm + lr;
      a.q[0] = *(const uint4*)&As[cur][row * 64 + ks * 32 + sel * 16];
      a.q[1] = *(const uint4*)&As[cur][row * 64 + ks * 32 + sel * 16 + 8];
#pragma unroll
      for (int j = 0; j < 4; ++j) {
        const int n = 64 * wn + 16 * j + lr;
        FragU b;
        const uint4* p = (const uint4*)&Bs[cur][n * 64 + ks * 32 + sel * 16];
        b.q[0] = p[0];
        b.q[1] = p[1];
        acc[j] = wmma_bf16(a.v, b.v, acc[j]);
      }
    }
    __syncthreads();
  }

  // epilogue: bias + relu -> bf16 E[16384][128]
#pragma unroll
  for (int j = 0; j < 4; ++j) {
    int col = 64 * wn + 16 * j + lr;
    float bv = bias[col];
#pragma unroll
    for (int r = 0; r < 8; ++r) {
      int row = m0 + 16 * wm + sel * 8 + r;
      float v = acc[j][r] + bv;
      v = v > 0.f ? v : 0.f;
      E[(size_t)row * 128 + col] = f2bf(v);
    }
  }
}

// ---------------------------------------------------------------------------
// K2: fused extract-layer2 + base learner, 128 rows / block, WMMA layers
// ping-pong inside one LDS buffer.
// ---------------------------------------------------------------------------
template <int K, int N, bool RELU>
__device__ __forceinline__ void layer_run(unsigned short* Ab,
                                          const unsigned short* Wb,
                                          const float* biasS) {
  const int tid = threadIdx.x;
  const int lane = tid & 31;
  const int wave = tid >> 5;
  const int lr = lane & 15;
  const int sel = lane >> 4;
  constexpr int NT = N / 16;
  constexpr int KT = K / 32;

  f32x8 acc[NT];
  const f32x8 vzero = {0.f, 0.f, 0.f, 0.f, 0.f, 0.f, 0.f, 0.f};
#pragma unroll
  for (int nt = 0; nt < NT; ++nt) acc[nt] = vzero;

  const int row = wave * 16 + lr;
#pragma unroll
  for (int kt = 0; kt < KT; ++kt) {
    const int k0 = kt * 32;
    FragU a;
    a.q[0] = *(const uint4*)&Ab[row * K + k0 + sel * 16];
    a.q[1] = *(const uint4*)&Ab[row * K + k0 + sel * 16 + 8];
#pragma unroll
    for (int nt = 0; nt < NT; ++nt) {
      int n = nt * 16 + lr;
      FragU b;
      const uint4* p = (const uint4*)&Wb[n * K + k0 + sel * 16];
      b.q[0] = p[0];
      b.q[1] = p[1];
      acc[nt] = wmma_bf16(a.v, b.v, acc[nt]);
    }
  }
  __syncthreads();  // all reads of Ab complete before overwriting
#pragma unroll
  for (int nt = 0; nt < NT; ++nt) {
    int col = nt * 16 + lr;
    float bv = biasS[col];
#pragma unroll
    for (int r = 0; r < 8; ++r) {
      int orow = wave * 16 + sel * 8 + r;
      float v = acc[nt][r] + bv;
      if (RELU) v = v > 0.f ? v : 0.f;
      Ab[orow * N + col] = f2bf(v);
    }
  }
  __syncthreads();
}

// weight -> bf16 n-major staging (no trailing barrier)
template <int K, int N>
__device__ __forceinline__ void stage_w_ns(const float* __restrict__ W,
                                           const float* __restrict__ b,
                                           unsigned short* Wb, float* biasS) {
  for (int i = threadIdx.x; i < K * N; i += 256) {
    int k = i / N;
    int n = i - k * N;
    Wb[n * K + k] = f2bf(W[i]);
  }
  if (threadIdx.x < N) biasS[threadIdx.x] = b[threadIdx.x];
}

__global__ void __launch_bounds__(256) tail_kernel(
    const unsigned short* __restrict__ Ee, const unsigned short* __restrict__ Ea,
    const float* w2e, const float* b2e, const float* w2a, const float* b2a,
    const float* w1be, const float* b1be, const float* w2be, const float* b2be,
    const float* w3be, const float* b3be,
    const float* w1ba, const float* b1ba, const float* w2ba, const float* b2ba,
    const float* w3ba, const float* b3ba,
    float* __restrict__ predE, float* __restrict__ predA) {
  const unsigned short* E = blockIdx.y ? Ea : Ee;
  const float* w2x = blockIdx.y ? w2a : w2e;
  const float* b2x = blockIdx.y ? b2a : b2e;
  const float* w1b = blockIdx.y ? w1ba : w1be;
  const float* b1b = blockIdx.y ? b1ba : b1be;
  const float* w2b = blockIdx.y ? w2ba : w2be;
  const float* b2b = blockIdx.y ? b2ba : b2be;
  const float* w3b = blockIdx.y ? w3ba : w3be;
  const float* b3b = blockIdx.y ? b3ba : b3be;
  float* pred = blockIdx.y ? predA : predE;

  __shared__ unsigned short Ab[128 * 128];  // 32 KB activation ping-pong
  __shared__ unsigned short Wb[128 * 64];   // 16 KB weights (bf16, n-major)
  __shared__ float biasS[128];
  __shared__ float w3S[128];
  __shared__ float b3S[2];

  const int tid = threadIdx.x;
  const int m0 = blockIdx.x * 128;

  // stage E rows (bf16) via async-to-LDS, overlapped with weight staging
  {
    int r = tid >> 1, h = tid & 1;
    async_copy128(E + (size_t)(m0 + r) * 128 + h * 64, &Ab[r * 128 + h * 64]);
  }
  stage_w_ns<128, 64>(w2x, b2x, Wb, biasS);
  wait_async0();
  __syncthreads();
  // layer 1: feat = E @ W2 + b2   (128 -> 64, no relu)
  layer_run<128, 64, false>(Ab, Wb, biasS);
  // layer 2: h1 = relu(feat @ w1b + b1b)   (64 -> 128)
  stage_w_ns<64, 128>(w1b, b1b, Wb, biasS);
  __syncthreads();
  layer_run<64, 128, true>(Ab, Wb, biasS);
  // layer 3: h2 = relu(h1 @ w2b + b2b)    (128 -> 64)
  stage_w_ns<128, 64>(w2b, b2b, Wb, biasS);
  __syncthreads();
  layer_run<128, 64, true>(Ab, Wb, biasS);
  // layer 4: logits (64 -> 2) + softmax  (VALU)
  if (tid < 128) w3S[tid] = w3b[tid];
  if (tid < 2) b3S[tid] = b3b[tid];
  __syncthreads();
  if (tid < 128) {
    float a0 = b3S[0], a1 = b3S[1];
#pragma unroll
    for (int k = 0; k < 64; ++k) {
      float v = bf2f(Ab[tid * 64 + k]);
      a0 += v * w3S[k * 2 + 0];
      a1 += v * w3S[k * 2 + 1];
    }
    float m = fmaxf(a0, a1);
    float e0 = __expf(a0 - m), e1 = __expf(a1 - m);
    float inv = 1.f / (e0 + e1);
    pred[(size_t)(m0 + tid) * 2 + 0] = e0 * inv;
    pred[(size_t)(m0 + tid) * 2 + 1] = e1 * inv;
  }
}

// ---------------------------------------------------------------------------
// K3: per-row fusion network (features -> single/comp MLPs -> mix -> softmax)
// ---------------------------------------------------------------------------
__global__ void __launch_bounds__(256) fuse_kernel(
    const float* __restrict__ predE, const float* __restrict__ predA,
    const int* __restrict__ labels,
    const float* sw1, const float* sb1, const float* sw2, const float* sb2,
    const float* cw1, const float* cb1, const float* cw2, const float* cb2,
    float* __restrict__ out) {
  __shared__ float S1[6 * 64], SB1[64], S2[64 * 4], SB2[4];
  __shared__ float C1[6 * 64], CB1[64], C2[64 * 2], CB2[2];
  const int tid = threadIdx.x;
  for (int i = tid; i < 384; i += 256) { S1[i] = sw1[i]; C1[i] = cw1[i]; }
  if (tid < 64) { SB1[tid] = sb1[tid]; CB1[tid] = cb1[tid]; }
  if (tid < 256) S2[tid] = sw2[tid];
  if (tid < 128) C2[tid] = cw2[tid];
  if (tid < 4) SB2[tid] = sb2[tid];
  if (tid < 2) CB2[tid] = cb2[tid];
  __syncthreads();

  const int row = blockIdx.x * 256 + tid;
  float p0[2], p1[2];
  p0[0] = predE[row * 2 + 0];
  p0[1] = predE[row * 2 + 1];
  p1[0] = predA[row * 2 + 0];
  p1[1] = predA[row * 2 + 1];
  const int lab = labels[row];
  float c0 = ((p0[1] > p0[0] ? 1 : 0) == lab) ? 1.f : 0.f;
  float c1 = ((p1[1] > p1[0] ? 1 : 0) == lab) ? 1.f : 0.f;
  float f[6] = {c0, p0[0], p0[1], c1, p1[0], p1[1]};

  float s[4] = {SB2[0], SB2[1], SB2[2], SB2[3]};
  float c[2] = {CB2[0], CB2[1]};
#pragma unroll 4
  for (int j = 0; j < 64; ++j) {
    float hs = SB1[j], hc = CB1[j];
#pragma unroll
    for (int i = 0; i < 6; ++i) {
      hs += f[i] * S1[i * 64 + j];
      hc += f[i] * C1[i * 64 + j];
    }
    hs = hs > 0.f ? hs : 0.f;
    hc = hc > 0.f ? hc : 0.f;
#pragma unroll
    for (int o = 0; o < 4; ++o) s[o] += hs * S2[j * 4 + o];
#pragma unroll
    for (int o = 0; o < 2; ++o) c[o] += hc * C2[j * 2 + o];
  }
#pragma unroll
  for (int o = 0; o < 4; ++o) s[o] = 1.f / (1.f + __expf(-s[o]));
#pragma unroll
  for (int o = 0; o < 2; ++o) c[o] = 1.f / (1.f + __expf(-c[o]));

  float res[2];
#pragma unroll
  for (int k = 0; k < 2; ++k) {
    float mu1 = s[k], mu2 = s[2 + k];
    float mu12 = fmaxf(mu1, mu2) + c[k];
    res[k] = (p0[k] <= p1[k]) ? p0[k] * mu1 + (p1[k] - p0[k]) * mu12
                              : p0[k] * mu2 + (p1[k] - p0[k]) * mu12;
  }
  float m = fmaxf(res[0], res[1]);
  float e0 = __expf(res[0] - m), e1 = __expf(res[1] - m);
  float inv = 1.f / (e0 + e1);
  out[row * 2 + 0] = e0 * inv;
  out[row * 2 + 1] = e1 * inv;
}

// ---------------------------------------------------------------------------
extern "C" void kernel_launch(void* const* d_in, const int* in_sizes, int n_in,
                              void* d_out, int out_size, void* d_ws,
                              size_t ws_size, hipStream_t stream) {
  const float* eeg = (const float*)d_in[0];
  const float* aud = (const float*)d_in[1];
  const int* labels = (const int*)d_in[2];
  const float* w1e = (const float*)d_in[3];
  const float* b1e = (const float*)d_in[4];
  const float* w2e = (const float*)d_in[5];
  const float* b2e = (const float*)d_in[6];
  const float* w1a = (const float*)d_in[7];
  const float* b1a = (const float*)d_in[8];
  const float* w2a = (const float*)d_in[9];
  const float* b2a = (const float*)d_in[10];
  const float* w1be = (const float*)d_in[11];
  const float* b1be = (const float*)d_in[12];
  const float* w2be = (const float*)d_in[13];
  const float* b2be = (const float*)d_in[14];
  const float* w3be = (const float*)d_in[15];
  const float* b3be = (const float*)d_in[16];
  const float* w1ba = (const float*)d_in[17];
  const float* b1ba = (const float*)d_in[18];
  const float* w2ba = (const float*)d_in[19];
  const float* b2ba = (const float*)d_in[20];
  const float* w3ba = (const float*)d_in[21];
  const float* b3ba = (const float*)d_in[22];
  const float* sw1 = (const float*)d_in[23];
  const float* sb1 = (const float*)d_in[24];
  const float* sw2 = (const float*)d_in[25];
  const float* sb2 = (const float*)d_in[26];
  const float* cw1 = (const float*)d_in[27];
  const float* cb1 = (const float*)d_in[28];
  const float* cw2 = (const float*)d_in[29];
  const float* cb2 = (const float*)d_in[30];

  char* ws = (char*)d_ws;
  unsigned short* W1T_E = (unsigned short*)(ws + 0);
  unsigned short* W1T_A = (unsigned short*)(ws + (1u << 20));
  unsigned short* E_E = (unsigned short*)(ws + (2u << 20));
  unsigned short* E_A = (unsigned short*)(ws + (6u << 20));
  float* PRED_E = (float*)(ws + (10u << 20));
  float* PRED_A = (float*)(ws + (10u << 20) + 16384u * 2u * 4u);

  wconv_kernel<<<dim3(2048, 2), 256, 0, stream>>>(w1e, w1a, W1T_E, W1T_A);
  gemm1_kernel<<<dim3(256, 2), 256, 0, stream>>>(eeg, aud, W1T_E, W1T_A, b1e,
                                                 b1a, E_E, E_A);
  tail_kernel<<<dim3(128, 2), 256, 0, stream>>>(
      E_E, E_A, w2e, b2e, w2a, b2a, w1be, b1be, w2be, b2be, w3be, b3be, w1ba,
      b1ba, w2ba, b2ba, w3ba, b3ba, PRED_E, PRED_A);
  fuse_kernel<<<dim3(64), 256, 0, stream>>>(PRED_E, PRED_A, labels, sw1, sb1,
                                            sw2, sb2, cw1, cb1, cw2, cb2,
                                            (float*)d_out);
}
